// MultiHeadSelfAttention_10247791968609
// MI455X (gfx1250) — matrix operand order repaired and checked by reading, not compiled, not run on hardware
//
#include <hip/hip_runtime.h>

// ---------------- problem constants ----------------
#define BB 4
#define SS 2048
#define DD 1024
#define HH 16
#define TT 64   // head dim

typedef __bf16 bf16;
typedef __attribute__((ext_vector_type(16))) __bf16 v16bf;
typedef __attribute__((ext_vector_type(8)))  __bf16 v8bf;
typedef __attribute__((ext_vector_type(8)))  float  v8f;

static __device__ inline v16bf cat16(v8bf lo, v8bf hi) {
  return __builtin_shufflevector(lo, hi, 0,1,2,3,4,5,6,7,8,9,10,11,12,13,14,15);
}
static __device__ inline v8f wmma_bf16(v16bf a, v16bf b, v8f c) {
  // D = A(16x32 bf16) * B(32x16 bf16) + C(16x16 f32)
  return __builtin_amdgcn_wmma_f32_16x16x32_bf16(false, a, false, b, (short)0, c, false, false);
}
static __device__ inline v8f vzero8() {
  v8f z;
  #pragma unroll
  for (int i = 0; i < 8; ++i) z[i] = 0.0f;
  return z;
}
static __device__ inline v8bf cvt8(const float4 a, const float4 b) {
  v8bf r;
  r[0] = (bf16)a.x; r[1] = (bf16)a.y; r[2] = (bf16)a.z; r[3] = (bf16)a.w;
  r[4] = (bf16)b.x; r[5] = (bf16)b.y; r[6] = (bf16)b.z; r[7] = (bf16)b.w;
  return r;
}
// Flat address of an LDS object: high 32 bits = shared aperture, low 32 bits
// = LDS offset (ISA 10.2 aperture mapping). Truncate to get the DS address.
static __device__ inline unsigned lds_u32(const void* p) {
  return (unsigned)(unsigned long long)p;
}
// CDNA5 async global->LDS copy (16B per lane), tracked with ASYNCcnt.
static __device__ inline void async_b128(unsigned lds_addr, const bf16* gptr) {
  asm volatile("global_load_async_to_lds_b128 %0, %1, off"
               :: "v"(lds_addr), "v"(gptr) : "memory");
}
static __device__ inline void wait_async0() {
  asm volatile("s_wait_asynccnt 0x0" ::: "memory");
}

// =====================================================================
// GEMM with bias: Y[m,n] = sum_k A[m,k] * W[n,k] + bias[n]
//   A: M x 1024 (f32 or bf16, row-major); W: 1024 x 1024 f32 (row = n)
// Block: 256 thr (8 waves). Tile: 128 rows x 128 cols. K-step 32.
// Wave w: rows 16w..16w+15, all 128 cols (8 accumulators).
// =====================================================================
template <bool A_IS_F32, bool OUT_F32>
__global__ __launch_bounds__(256)
void gemm_bias_kernel(const void* __restrict__ Ap,
                      const float* __restrict__ W,
                      const float* __restrict__ bias,
                      void* __restrict__ Yp)
{
  const int t    = threadIdx.x;
  const int w    = t >> 5;
  const int lane = t & 31;
  const int half = lane >> 4;
  const int l16  = lane & 15;
  const int rowBlock = blockIdx.x * 128;
  const int colBlock = blockIdx.y * 128;

  __shared__ alignas(16) bf16 Al[128][32];
  __shared__ alignas(16) bf16 Wl[128][32];

  v8f acc[8];
  #pragma unroll
  for (int i = 0; i < 8; ++i) acc[i] = vzero8();

  const int ar = t >> 1;          // 0..127
  const int ac = (t & 1) * 16;    // 0 or 16

  for (int k0 = 0; k0 < DD; k0 += 32) {
    // ---- cooperative load A tile (128x32) -> bf16 LDS ----
    if (A_IS_F32) {
      const float4* s4 = (const float4*)((const float*)Ap + (size_t)(rowBlock + ar) * DD + k0 + ac);
      *(v8bf*)&Al[ar][ac]     = cvt8(s4[0], s4[1]);
      *(v8bf*)&Al[ar][ac + 8] = cvt8(s4[2], s4[3]);
    } else {
      const v8bf* s8 = (const v8bf*)((const bf16*)Ap + (size_t)(rowBlock + ar) * DD + k0 + ac);
      *(v8bf*)&Al[ar][ac]     = s8[0];
      *(v8bf*)&Al[ar][ac + 8] = s8[1];
    }
    // ---- cooperative load W tile (128x32) -> bf16 LDS ----
    {
      const float4* s4 = (const float4*)(W + (size_t)(colBlock + ar) * DD + k0 + ac);
      *(v8bf*)&Wl[ar][ac]     = cvt8(s4[0], s4[1]);
      *(v8bf*)&Wl[ar][ac + 8] = cvt8(s4[2], s4[3]);
    }
    __syncthreads();

    // ---- A fragment: rows w*16 + l16, K = k0..k0+31 ----
    v8bf alo = *(const v8bf*)&Al[w * 16 + l16][half * 16];
    v8bf ahi = *(const v8bf*)&Al[w * 16 + l16][half * 16 + 8];
    v16bf afrag = cat16(alo, ahi);

    // ---- 8 B fragments, 8 WMMAs ----
    #pragma unroll
    for (int nt = 0; nt < 8; ++nt) {
      v8bf blo = *(const v8bf*)&Wl[nt * 16 + l16][half * 16];
      v8bf bhi = *(const v8bf*)&Wl[nt * 16 + l16][half * 16 + 8];
      acc[nt] = wmma_bf16(afrag, cat16(blo, bhi), acc[nt]);
    }
    __syncthreads();
  }

  // ---- epilogue: C/D layout row = r + 8*half, col = l16 ----
  #pragma unroll
  for (int nt = 0; nt < 8; ++nt) {
    const int col = colBlock + nt * 16 + l16;
    const float bv = bias[col];
    #pragma unroll
    for (int r = 0; r < 8; ++r) {
      const int row = rowBlock + w * 16 + half * 8 + r;
      const float v = acc[nt][r] + bv;
      if (OUT_F32) ((float*)Yp)[(size_t)row * DD + col] = v;
      else         ((bf16*)Yp)[(size_t)row * DD + col] = (bf16)v;
    }
  }
}

// =====================================================================
// Flash attention per (b, h, 128-query block).
// 8 waves x 16 query rows. Key blocks of 64. Online softmax.
// Row sums via WMMA against a ones matrix; only row-max uses shuffles.
// =====================================================================
__global__ __launch_bounds__(256)
void attn_kernel(const bf16* __restrict__ qb,
                 const bf16* __restrict__ kb,
                 const bf16* __restrict__ vb,
                 bf16* __restrict__ zb)
{
  const int t    = threadIdx.x;
  const int w    = t >> 5;
  const int lane = t & 31;
  const int half = lane >> 4;
  const int l16  = lane & 15;
  const int h    = blockIdx.y;
  const int b    = blockIdx.z;
  const size_t headOff = (size_t)b * SS * DD + (size_t)h * TT;

  __shared__ alignas(16) bf16 Kl[64][72];     // [key][t]   (pitch 72 vs banks)
  __shared__ alignas(16) bf16 Vt[64][72];     // [t][key]   (transposed)
  __shared__ alignas(16) bf16 Pl[8][16][64];  // per-wave P scratch [row][key]

  // ---- load Q fragments once: rows qrow, contraction K = t-dim (64) ----
  const int qrowIdx = blockIdx.x * 128 + w * 16 + l16;
  const bf16* qrow = qb + headOff + (size_t)qrowIdx * DD;
  v16bf qf[2];
  #pragma unroll
  for (int kc = 0; kc < 2; ++kc) {
    v8bf lo = *(const v8bf*)(qrow + kc * 32 + half * 16);
    v8bf hi = *(const v8bf*)(qrow + kc * 32 + half * 16 + 8);
    qf[kc] = cat16(lo, hi);
  }

  v16bf ones;
  #pragma unroll
  for (int i = 0; i < 16; ++i) ones[i] = (bf16)1.0f;

  v8f o[4];
  #pragma unroll
  for (int i = 0; i < 4; ++i) o[i] = vzero8();
  float mrow[8], lrow[8];
  #pragma unroll
  for (int r = 0; r < 8; ++r) { mrow[r] = -3.0e38f; lrow[r] = 0.0f; }

  const float sc = 0.125f * 1.44269504088896f; // (1/sqrt(T)) * log2(e)

  const int kr   = t >> 2;        // staging: key row 0..63
  const int kc16 = (t & 3) * 16;  // staging: t-dim offset 0/16/32/48

  for (int j = 0; j < SS; j += 64) {
    // ---- stage K via async global->LDS; V manually transposed ----
    {
      const bf16* ksrc = kb + headOff + (size_t)(j + kr) * DD + kc16;
      async_b128(lds_u32(&Kl[kr][kc16]),     ksrc);
      async_b128(lds_u32(&Kl[kr][kc16 + 8]), ksrc + 8);
      const bf16* vsrc = vb + headOff + (size_t)(j + kr) * DD + kc16;
      v8bf v0 = *(const v8bf*)vsrc;
      v8bf v1 = *(const v8bf*)(vsrc + 8);
      #pragma unroll
      for (int i = 0; i < 8; ++i) {
        Vt[kc16 + i][kr]     = v0[i];
        Vt[kc16 + 8 + i][kr] = v1[i];
      }
      wait_async0();
    }
    __syncthreads();

    // ---- scores: S(16x64 keys) = q(16x64) * K^T ----
    v8f s[4];
    #pragma unroll
    for (int i = 0; i < 4; ++i) s[i] = vzero8();
    #pragma unroll
    for (int kc = 0; kc < 2; ++kc) {
      #pragma unroll
      for (int kt = 0; kt < 4; ++kt) {  // key tile: keys kt*16..kt*16+15
        v8bf blo = *(const v8bf*)&Kl[kt * 16 + l16][kc * 32 + half * 16];
        v8bf bhi = *(const v8bf*)&Kl[kt * 16 + l16][kc * 32 + half * 16 + 8];
        s[kt] = wmma_bf16(qf[kc], cat16(blo, bhi), s[kt]);
      }
    }

    // ---- online softmax (max via 16-lane shuffles, sum via ones-WMMA) ----
    float al[8];
    #pragma unroll
    for (int r = 0; r < 8; ++r) {
      float x0 = s[0][r], x1 = s[1][r], x2 = s[2][r], x3 = s[3][r];
      float mx = fmaxf(fmaxf(x0, x1), fmaxf(x2, x3));
      #pragma unroll
      for (int msk = 1; msk <= 8; msk <<= 1) mx = fmaxf(mx, __shfl_xor(mx, msk, 32));
      const float mnew = fmaxf(mrow[r], mx);
      al[r] = exp2f((mrow[r] - mnew) * sc);
      mrow[r] = mnew;
      const float p0 = exp2f((x0 - mnew) * sc);
      const float p1 = exp2f((x1 - mnew) * sc);
      const float p2 = exp2f((x2 - mnew) * sc);
      const float p3 = exp2f((x3 - mnew) * sc);
      #pragma unroll
      for (int nt = 0; nt < 4; ++nt) o[nt][r] *= al[r];
      const int prw = r + 8 * half;
      Pl[w][prw][l16]      = (bf16)p0;
      Pl[w][prw][16 + l16] = (bf16)p1;
      Pl[w][prw][32 + l16] = (bf16)p2;
      Pl[w][prw][48 + l16] = (bf16)p3;
    }

    // ---- P fragments (per-wave LDS; same-wave DS ordering suffices) ----
    v16bf pf[2];
    #pragma unroll
    for (int kc = 0; kc < 2; ++kc) {
      v8bf plo = *(const v8bf*)&Pl[w][l16][kc * 32 + half * 16];
      v8bf phi = *(const v8bf*)&Pl[w][l16][kc * 32 + half * 16 + 8];
      pf[kc] = cat16(plo, phi);
    }

    // ---- row sums: ls = P * ones (broadcast across cols) ----
    v8f ls = vzero8();
    ls = wmma_bf16(pf[0], ones, ls);
    ls = wmma_bf16(pf[1], ones, ls);
    #pragma unroll
    for (int r = 0; r < 8; ++r) lrow[r] = lrow[r] * al[r] + ls[r];

    // ---- ctx: O += P(16x64) * V(64x64) ----
    #pragma unroll
    for (int kc = 0; kc < 2; ++kc) {
      #pragma unroll
      for (int nt = 0; nt < 4; ++nt) {
        v8bf vlo = *(const v8bf*)&Vt[nt * 16 + l16][kc * 32 + half * 16];
        v8bf vhi = *(const v8bf*)&Vt[nt * 16 + l16][kc * 32 + half * 16 + 8];
        o[nt] = wmma_bf16(pf[kc], cat16(vlo, vhi), o[nt]);
      }
    }
    __syncthreads();
  }

  // ---- normalize and write z (concat-head layout) ----
  #pragma unroll
  for (int r = 0; r < 8; ++r) {
    const float inv = 1.0f / lrow[r];
    const int row = blockIdx.x * 128 + w * 16 + half * 8 + r;
    #pragma unroll
    for (int nt = 0; nt < 4; ++nt) {
      zb[headOff + (size_t)row * DD + nt * 16 + l16] = (bf16)(o[nt][r] * inv);
    }
  }
}

// =====================================================================
// launcher
// =====================================================================
extern "C" void kernel_launch(void* const* d_in, const int* in_sizes, int n_in,
                              void* d_out, int out_size, void* d_ws, size_t ws_size,
                              hipStream_t stream) {
  const float* Q  = (const float*)d_in[0];
  const float* K  = (const float*)d_in[1];
  const float* V  = (const float*)d_in[2];
  const float* Wq = (const float*)d_in[3];
  const float* bq = (const float*)d_in[4];
  const float* Wk = (const float*)d_in[5];
  const float* bk = (const float*)d_in[6];
  const float* Wv = (const float*)d_in[7];
  const float* bv = (const float*)d_in[8];
  const float* Wo = (const float*)d_in[9];
  const float* bo = (const float*)d_in[10];

  const size_t elems = (size_t)BB * SS * DD; // 8M elements per bf16 buffer
  bf16* qb = (bf16*)d_ws;
  bf16* kb = qb + elems;
  bf16* vb = kb + elems;
  bf16* zb = vb + elems;

  const dim3 gGemm(BB * SS / 128, DD / 128); // 64 x 8
  const dim3 blk(256);

  // QKV projections (f32 in, bf16 out)
  gemm_bias_kernel<true, false><<<gGemm, blk, 0, stream>>>(Q, Wq, bq, qb);
  gemm_bias_kernel<true, false><<<gGemm, blk, 0, stream>>>(K, Wk, bk, kb);
  gemm_bias_kernel<true, false><<<gGemm, blk, 0, stream>>>(V, Wv, bv, vb);

  // attention
  const dim3 gAttn(SS / 128, HH, BB); // 16 x 16 x 4
  attn_kernel<<<gAttn, blk, 0, stream>>>(qb, kb, vb, zb);

  // output projection (bf16 in, f32 out)
  gemm_bias_kernel<false, true><<<gGemm, blk, 0, stream>>>(zb, Wo, bo, d_out);
}
